// BRFEGNN_56642028700085
// MI455X (gfx1250) — matrix operand order repaired and checked by reading, not compiled, not run on hardware
//
#include <hip/hip_runtime.h>
#include <hip/hip_bf16.h>

#define DIN 512
#define HD  128
#define NC  40

typedef __attribute__((ext_vector_type(16))) __bf16 v16bf;
typedef __attribute__((ext_vector_type(8)))  float  v8f;

union AB16 {
    v16bf        v;
    uint4        q[2];
    unsigned int w[8];
    unsigned short u[16];
};

__device__ __forceinline__ unsigned int f2bf(float f) {
    unsigned int u = __float_as_uint(f);
    u += 0x7FFFu + ((u >> 16) & 1u);      // round-to-nearest-even
    return (u >> 16);
}

// ---------------- degree / normalization ----------------
__global__ void k_deg_init(float* deg, int n) {
    int i = blockIdx.x * blockDim.x + threadIdx.x;
    if (i < n) deg[i] = 1.0f;                      // self-loop contribution
}

__global__ void k_deg_edges(const int* __restrict__ ei, float* deg, int E) {
    int e = blockIdx.x * blockDim.x + threadIdx.x;
    if (e < E) atomicAdd(&deg[ei[E + e]], 1.0f);   // dst row of edge_idx
}

__global__ void k_dinv(float* deg, int n) {
    int i = blockIdx.x * blockDim.x + threadIdx.x;
    if (i < n) deg[i] = rsqrtf(deg[i]);            // deg >= 1 always
}

// ---------------- pack W[512,128] f32 -> bf16 in per-lane WMMA B layout ------
// Packed unit index: (((kt*8 + wave)*32 + lane)*8 + i), kt = K-tile of 32.
// Value: bf16 pair W[k][col], W[k+1][col] with k = kt*32 + 16*(lane>>4) + 2i,
// col = wave*16 + (lane&15).  Lane's 8 dwords are contiguous (32B) -> 2x b128.
__global__ void k_pack_w(const float* __restrict__ W, unsigned int* __restrict__ Wpk)
{
    int t = blockIdx.x * blockDim.x + threadIdx.x;          // [0, 16*8*32*8)
    if (t >= (DIN / 32) * 8 * 32 * 8) return;
    int i    = t & 7;
    int lane = (t >> 3) & 31;
    int w    = (t >> 8) & 7;
    int kt   = t >> 11;
    int k    = kt * 32 + 16 * (lane >> 4) + 2 * i;
    int col  = w * 16 + (lane & 15);
    unsigned int lo = f2bf(W[(size_t)k       * HD + col]);
    unsigned int hi = f2bf(W[(size_t)(k + 1) * HD + col]);
    Wpk[t] = lo | (hi << 16);
}

// ---------------- main GEMM: H[n,128] = X[n,512] @ W[512,128] (bf16 WMMA) ----
__global__ void __launch_bounds__(256)
k_gemm_bf16(const float* __restrict__ X, const unsigned int* __restrict__ Wpk,
            float* __restrict__ H, int n)
{
    __shared__ unsigned short lds_a[16 * 64];      // 16 rows x 64 K (bf16), 2KB
    const int m0      = blockIdx.x * 16;
    const int wave    = threadIdx.x >> 5;
    const int lane    = threadIdx.x & 31;
    const int half    = lane >> 4;
    const int l15     = lane & 15;
    const int colbase = wave * 16;                 // 8 waves cover 128 cols

    v8f acc = {};

    for (int k0 = 0; k0 < DIN; k0 += 64) {
        __syncthreads();
        // stage 16x64 f32 A tile -> bf16 LDS; 512 dwords, 2 float2 per thread
#pragma unroll
        for (int j = 0; j < 2; ++j) {
            int d  = threadIdx.x + j * 256;        // dword index in tile
            int r  = d >> 5;                       // row (32 dwords per row)
            int kd = d & 31;                       // dword within row
            float2 xv = make_float2(0.0f, 0.0f);
            if (m0 + r < n)
                xv = *(const float2*)&X[(size_t)(m0 + r) * DIN + k0 + kd * 2];
            ((unsigned int*)lds_a)[d] = f2bf(xv.x) | (f2bf(xv.y) << 16);
        }
        __syncthreads();

#pragma unroll
        for (int sub = 0; sub < 2; ++sub) {
            // A per-lane layout: lanes 0-15 -> K{0..15}, 16-31 -> K{16..31}
            AB16 a;
            const unsigned short* ap = &lds_a[l15 * 64 + sub * 32 + 16 * half];
            a.q[0] = *(const uint4*)ap;            // 16B-aligned ds_load_b128
            a.q[1] = *(const uint4*)(ap + 8);

            AB16 b;                                // pre-packed, 32B contiguous
            int kt = (k0 >> 5) + sub;
            const uint4* bp =
                (const uint4*)&Wpk[((size_t)(kt * 8 + wave) * 32 + lane) * 8];
            b.q[0] = bp[0];
            b.q[1] = bp[1];

            acc = __builtin_amdgcn_wmma_f32_16x16x32_bf16(
                false, a.v, false, b.v, (short)0, acc, false, false);
        }
    }

    // C layout: VGPR r -> row (r + 8*half), col l15
#pragma unroll
    for (int r = 0; r < 8; ++r) {
        int row = m0 + r + 8 * half;
        if (row < n) H[(size_t)row * HD + colbase + l15] = acc[r];
    }
}

// ---------------- layer-1 aggregation ----------------
__global__ void k_self128(const float* __restrict__ ha, const float* __restrict__ hb,
                          const float* __restrict__ dinv,
                          float* agg0, float* agg1, int n)
{
    int t = blockIdx.x * blockDim.x + threadIdx.x;
    if (t < n * HD) {
        int node = t >> 7;
        float c = dinv[node]; c *= c;              // self-loop coef
        agg0[t] = ha[t] * c;
        agg1[t] = hb[t] * c;
    }
}

__global__ void __launch_bounds__(256)
k_scatter128(const int* __restrict__ ei, const float* __restrict__ dinv,
             const float* __restrict__ ha, const float* __restrict__ hb,
             float* agg0, float* agg1, int E)
{
    int wave = threadIdx.x >> 5;
    int lane = threadIdx.x & 31;
    int e = blockIdx.x * 8 + wave;                 // one wave32 per edge
    if (e >= E) return;
    int s = ei[e], d = ei[E + e];
    float c = dinv[s] * dinv[d];
    int f = lane * 4;                              // 32 lanes x float4 = 128 feats
    float4 va = *(const float4*)&ha[(size_t)s * HD + f];
    atomicAdd(&agg0[(size_t)d * HD + f + 0], va.x * c);
    atomicAdd(&agg0[(size_t)d * HD + f + 1], va.y * c);
    atomicAdd(&agg0[(size_t)d * HD + f + 2], va.z * c);
    atomicAdd(&agg0[(size_t)d * HD + f + 3], va.w * c);
    float4 vb = *(const float4*)&hb[(size_t)s * HD + f];
    atomicAdd(&agg1[(size_t)d * HD + f + 0], vb.x * c);
    atomicAdd(&agg1[(size_t)d * HD + f + 1], vb.y * c);
    atomicAdd(&agg1[(size_t)d * HD + f + 2], vb.z * c);
    atomicAdd(&agg1[(size_t)d * HD + f + 3], vb.w * c);
}

// ---------------- layer 2: z = relu(agg + b1) concat @ W2 ----------------
__global__ void k_lin2(const float* __restrict__ agg0, const float* __restrict__ agg1,
                       const float* __restrict__ b1a, const float* __restrict__ b1b,
                       const float* __restrict__ W2, float* z, int n)
{
    int t = blockIdx.x * blockDim.x + threadIdx.x;
    if (t >= n * NC) return;
    int node = t / NC, c = t - node * NC;
    const float* r0 = &agg0[(size_t)node * HD];
    const float* r1 = &agg1[(size_t)node * HD];
    float acc = 0.0f;
    for (int k = 0; k < HD; ++k) {
        float h = fmaxf(r0[k] + b1a[k], 0.0f);
        acc = fmaf(h, W2[(size_t)k * NC + c], acc);
    }
    for (int k = 0; k < HD; ++k) {
        float h = fmaxf(r1[k] + b1b[k], 0.0f);
        acc = fmaf(h, W2[(size_t)(HD + k) * NC + c], acc);
    }
    z[t] = acc;
}

__global__ void k_self40(const float* __restrict__ z, const float* __restrict__ dinv,
                         float* agg2, int n)
{
    int t = blockIdx.x * blockDim.x + threadIdx.x;
    if (t < n * NC) {
        int node = t / NC;
        float c = dinv[node]; c *= c;
        agg2[t] = z[t] * c;
    }
}

__global__ void k_scatter40(const int* __restrict__ ei, const float* __restrict__ dinv,
                            const float* __restrict__ z, float* agg2, int E)
{
    int t = blockIdx.x * blockDim.x + threadIdx.x;
    if (t >= E * NC) return;
    int e = t / NC, c = t - e * NC;
    int s = ei[e], d = ei[E + e];
    float co = dinv[s] * dinv[d];
    atomicAdd(&agg2[(size_t)d * NC + c], z[(size_t)s * NC + c] * co);
}

__global__ void k_logsoftmax(const float* __restrict__ agg2, const float* __restrict__ b2,
                             float* out, int n)
{
    int i = blockIdx.x * blockDim.x + threadIdx.x;
    if (i >= n) return;
    float v[NC];
    float m = -3.0e38f;
    for (int c = 0; c < NC; ++c) {
        v[c] = agg2[(size_t)i * NC + c] + b2[c];
        m = fmaxf(m, v[c]);
    }
    float s = 0.0f;
    for (int c = 0; c < NC; ++c) s += __expf(v[c] - m);
    float lse = __logf(s) + m;
    for (int c = 0; c < NC; ++c) out[(size_t)i * NC + c] = v[c] - lse;
}

extern "C" void kernel_launch(void* const* d_in, const int* in_sizes, int n_in,
                              void* d_out, int out_size, void* d_ws, size_t ws_size,
                              hipStream_t stream)
{
    const float* x0  = (const float*)d_in[0];
    const float* x1  = (const float*)d_in[1];
    const int*   ei  = (const int*)d_in[2];   // [2,E]: row0=src, row1=dst
    const float* W1a = (const float*)d_in[3];
    const float* b1a = (const float*)d_in[4];
    const float* W1b = (const float*)d_in[5];
    const float* b1b = (const float*)d_in[6];
    const float* W2  = (const float*)d_in[7];
    const float* b2  = (const float*)d_in[8];
    float* out = (float*)d_out;

    const int n = in_sizes[0] / DIN;
    const int E = in_sizes[2] / 2;
    const int WPK = (DIN / 32) * 8 * 32 * 8;  // packed-W dwords per matrix

    // workspace layout (floats): dinv | ha | hb | agg0 | agg1 | z | agg2 | wpk0 | wpk1
    float* ws   = (float*)d_ws;
    float* dinv = ws;
    float* ha   = dinv + n;
    float* hb   = ha + (size_t)n * HD;
    float* agg0 = hb + (size_t)n * HD;
    float* agg1 = agg0 + (size_t)n * HD;
    float* z    = agg1 + (size_t)n * HD;
    float* agg2 = z + (size_t)n * NC;
    unsigned int* wpk0 = (unsigned int*)(agg2 + (size_t)n * NC);
    unsigned int* wpk1 = wpk0 + WPK;

    dim3 blk(256);

    k_deg_init <<<(n + 255) / 256, blk, 0, stream>>>(dinv, n);
    k_deg_edges<<<(E + 255) / 256, blk, 0, stream>>>(ei, dinv, E);
    k_dinv     <<<(n + 255) / 256, blk, 0, stream>>>(dinv, n);

    k_pack_w<<<(WPK + 255) / 256, blk, 0, stream>>>(W1a, wpk0);
    k_pack_w<<<(WPK + 255) / 256, blk, 0, stream>>>(W1b, wpk1);

    int mblocks = (n + 15) / 16;
    k_gemm_bf16<<<mblocks, blk, 0, stream>>>(x0, wpk0, ha, n);
    k_gemm_bf16<<<mblocks, blk, 0, stream>>>(x1, wpk1, hb, n);

    k_self128   <<<((n * HD) + 255) / 256, blk, 0, stream>>>(ha, hb, dinv, agg0, agg1, n);
    k_scatter128<<<(E + 7) / 8,            blk, 0, stream>>>(ei, dinv, ha, hb, agg0, agg1, E);

    k_lin2      <<<((n * NC) + 255) / 256, blk, 0, stream>>>(agg0, agg1, b1a, b1b, W2, z, n);
    k_self40    <<<((n * NC) + 255) / 256, blk, 0, stream>>>(z, dinv, agg2, n);
    k_scatter40 <<<((E * NC) + 255) / 256, blk, 0, stream>>>(ei, dinv, z, agg2, E);

    k_logsoftmax<<<(n + 255) / 256,        blk, 0, stream>>>(agg2, b2, out, n);
}
